// CrossAttention_11613591569142
// MI455X (gfx1250) — matrix operand order repaired and checked by reading, not compiled, not run on hardware
//
#include <hip/hip_runtime.h>
#include <hip/hip_bf16.h>

// Problem sizes (fixed by the reference)
#define BB 8
#define NQS 2048
#define NKS 2048
#define DDIM 256

typedef _Float16 v16h __attribute__((ext_vector_type(16)));
typedef _Float16 v8h  __attribute__((ext_vector_type(8)));
typedef float    v8f  __attribute__((ext_vector_type(8)));
typedef unsigned int u32x4 __attribute__((ext_vector_type(4)));
typedef int i32x8 __attribute__((ext_vector_type(8)));
typedef int i32x4 __attribute__((ext_vector_type(4)));

union HU16 { v16h v; v8h h[2]; };

static __device__ __forceinline__ v8f wmma16(v16h a, v16h b, v8f c) {
  // D = A(16x32 f16) * B(32x16 f16) + C(16x16 f32)
  return __builtin_amdgcn_wmma_f32_16x16x32_f16(false, a, false, b, (short)0, c,
                                                false, false);
}

// ---------------------------------------------------------------------------
// f32 -> f16 elementwise convert
// ---------------------------------------------------------------------------
__global__ void cvt_f32_f16_kernel(const float* __restrict__ src,
                                   _Float16* __restrict__ dst, int n) {
  int i = blockIdx.x * blockDim.x + threadIdx.x;
  if (i < n) dst[i] = (_Float16)src[i];
}

// ---------------------------------------------------------------------------
// V^T producer: vt[b][d][n] = key_f16[b][n][d]   (LDS tiled transpose)
// ---------------------------------------------------------------------------
__global__ void transpose_v_kernel(const _Float16* __restrict__ kf,
                                   _Float16* __restrict__ vt) {
  __shared__ _Float16 tile[32][33];
  const int b = blockIdx.z;
  const int n0 = blockIdx.x * 32;
  const int d0 = blockIdx.y * 32;
  const _Float16* src = kf + (size_t)b * NKS * DDIM;
  tile[threadIdx.y][threadIdx.x] =
      src[(size_t)(n0 + threadIdx.y) * DDIM + d0 + threadIdx.x];
  __syncthreads();
  vt[(size_t)b * DDIM * NKS + (size_t)(d0 + threadIdx.y) * NKS + n0 +
     threadIdx.x] = tile[threadIdx.x][threadIdx.y];
}

// ---------------------------------------------------------------------------
// DFT (real part of FFT over seq axis) as a WMMA GEMM:
//   QF[b] = Cos(2048x2048) @ query[b](2048x256), Cos[r,m]=cos(2*pi*r*m/2048)
// 8 waves = 4 row-groups x 2 col-groups; acc[8] per wave (no spills).
// B operands are double-buffered so the next ds_load overlaps the WMMA.
// ---------------------------------------------------------------------------
__global__ void dft_kernel(const float* __restrict__ query,
                           _Float16* __restrict__ qf) {
  __shared__ _Float16 Xt[DDIM][40];       // [d][k], k-block of 32, padded

  const int tid  = threadIdx.x;
  const int b    = blockIdx.y;
  const int mt   = blockIdx.x;            // 64-row tile of output rows
  const int wave = tid >> 5;
  const int lane = tid & 31;
  const int lo   = lane & 15;
  const int hiW  = (lane >= 16) ? 1 : 0;
  const int h8   = hiW ? 8 : 0;
  const int rg   = wave & 3;              // row group (0..3)
  const int cg   = wave >> 2;             // col group (0..1)
  const int rbase = mt * 64 + rg * 16;
  const int r     = rbase + lo;           // row this lane supplies to A

  v8f acc[8];
#pragma unroll
  for (int j = 0; j < 8; ++j) acc[j] = (v8f){};

  const float* X = query + (size_t)b * NQS * DDIM;
  const float w0 = 6.283185307179586f / 2048.0f;

  for (int kk = 0; kk < 64; ++kk) {
    __syncthreads();                      // previous block fully consumed
    // ---- cooperative stage: 32x256 f32 -> Xt[d][k] f16 (transposed)
#pragma unroll
    for (int it = 0; it < 8; ++it) {
      const int e  = it * 256 + tid;      // float4 index in the 32x256 tile
      const int k  = e >> 6;              // 64 float4 per k-row
      const int d4 = (e & 63) * 4;
      const float4 xv =
          *(const float4*)&X[(size_t)(kk * 32 + k) * DDIM + d4];
      Xt[d4 + 0][k] = (_Float16)xv.x;
      Xt[d4 + 1][k] = (_Float16)xv.y;
      Xt[d4 + 2][k] = (_Float16)xv.z;
      Xt[d4 + 3][k] = (_Float16)xv.w;
    }
    __syncthreads();                      // block visible to all waves

    // ---- A tile: lanes<16 hold K = kk*32+{0..15}; lanes>=16: {16..31}
    const int k0 = kk * 32 + (hiW ? 16 : 0);
    const int k1 = kk * 32 + (hiW ? 24 : 8);
    v16h a;
#pragma unroll
    for (int i = 0; i < 8; ++i) {
      unsigned p0 = ((unsigned)r * (unsigned)(k0 + i)) & 2047u;
      unsigned p1 = ((unsigned)r * (unsigned)(k1 + i)) & 2047u;
      a[i]     = (_Float16)__cosf((float)p0 * w0);
      a[8 + i] = (_Float16)__cosf((float)p1 * w0);
    }

    // ---- B from LDS, double-buffered so load(j+1) overlaps wmma(j)
    const int kb = hiW ? 16 : 0;
    HU16 bcur, bnext;
    bcur.h[0] = *(const v8h*)&Xt[cg * 128 + lo][kb];
    bcur.h[1] = *(const v8h*)&Xt[cg * 128 + lo][kb + 8];
#pragma unroll
    for (int j = 0; j < 8; ++j) {
      if (j < 7) {
        const int d = cg * 128 + (j + 1) * 16 + lo;
        bnext.h[0] = *(const v8h*)&Xt[d][kb];
        bnext.h[1] = *(const v8h*)&Xt[d][kb + 8];
      }
      acc[j] = wmma16(a, bcur.v, acc[j]);
      if (j < 7) bcur = bnext;
    }
  }

  _Float16* out = qf + (size_t)b * NQS * DDIM;
#pragma unroll
  for (int j = 0; j < 8; ++j)
#pragma unroll
    for (int i = 0; i < 8; ++i)
      out[(size_t)(rbase + i + h8) * DDIM + cg * 128 + j * 16 + lo] =
          (_Float16)acc[j][i];
}

// ---------------------------------------------------------------------------
// Projection GEMM: out[m, n] = sum_c A[m,c] * W[n,c] + bias[n]   (torch Linear)
// 4 row-groups x 2 col-groups per workgroup; acc[8] per wave; W operands
// double-buffered to overlap global loads with WMMA.
// ---------------------------------------------------------------------------
__global__ void proj_kernel(const _Float16* __restrict__ A,
                            const _Float16* __restrict__ W,
                            const float* __restrict__ bias,
                            _Float16* __restrict__ out) {
  const int wave = threadIdx.x >> 5;
  const int lane = threadIdx.x & 31;
  const int lo   = lane & 15;
  const int hiW  = (lane >= 16) ? 1 : 0;
  const int h8   = hiW ? 8 : 0;
  const int rg   = wave & 3;
  const int cg   = wave >> 2;
  const int rbase = blockIdx.x * 64 + rg * 16;

  if (lane == 0) __builtin_prefetch(W, 0, 3);   // global_prefetch_b8

  v8f acc[8];
#pragma unroll
  for (int j = 0; j < 8; ++j) acc[j] = (v8f){};

  const _Float16* arow = A + (size_t)(rbase + lo) * DDIM;
  const _Float16* wrow = W + (size_t)(cg * 128 + lo) * DDIM;  // + j*16 rows
#pragma unroll
  for (int kk = 0; kk < 8; ++kk) {
    HU16 au;
    au.h[0] = *(const v8h*)(arow + kk * 32 + (hiW ? 16 : 0));
    au.h[1] = *(const v8h*)(arow + kk * 32 + (hiW ? 24 : 8));
    const int co = kk * 32 + (hiW ? 16 : 0);
    v16h bcur, bnext;
    bcur = *(const v16h*)(wrow + co);
#pragma unroll
    for (int j = 0; j < 8; ++j) {
      if (j < 7)
        bnext = *(const v16h*)(wrow + (size_t)(j + 1) * 16 * DDIM + co);
      acc[j] = wmma16(au.v, bcur, acc[j]);
      if (j < 7) bcur = bnext;
    }
  }

#pragma unroll
  for (int j = 0; j < 8; ++j) {
    const float bb = bias[cg * 128 + j * 16 + lo];
#pragma unroll
    for (int i = 0; i < 8; ++i)
      out[(size_t)(rbase + i + h8) * DDIM + cg * 128 + j * 16 + lo] =
          (_Float16)(acc[j][i] + bb);
  }
}

// ---------------------------------------------------------------------------
// Tensor Data Mover: 2D tile load into LDS (data_size = 2 bytes).
// Packs D# group0/group1 per CDNA5 ISA 8.3/8.4; groups 2/3 zero (2D tile).
// ---------------------------------------------------------------------------
static __device__ __forceinline__ void tdm_load_2d(
    unsigned lds_byte, unsigned long long gaddr, unsigned tile_d0,
    unsigned tile_d1, unsigned tensor_d0, unsigned tensor_d1,
    unsigned long long stride0) {
  u32x4 g0;
  g0[0] = 1u;                                             // count=1, user D#
  g0[1] = lds_byte;                                       // lds_addr
  g0[2] = (unsigned)(gaddr & 0xFFFFFFFFull);              // global_addr lo
  g0[3] = (unsigned)((gaddr >> 32) & 0x1FFFFFFull) | (2u << 30); // hi | type=2
  i32x8 g1;
  g1[0] = (int)(1u << 16);                                // data_size = 2B
  g1[1] = (int)((tensor_d0 & 0xFFFFu) << 16);             // td0[15:0]
  g1[2] = (int)((tensor_d0 >> 16) | ((tensor_d1 & 0xFFFFu) << 16));
  g1[3] = (int)((tensor_d1 >> 16) | (tile_d0 << 16));
  g1[4] = (int)(tile_d1 & 0xFFFFu);                       // tile_dim1, dim2=0
  g1[5] = (int)(stride0 & 0xFFFFFFFFull);
  g1[6] = (int)((stride0 >> 32) & 0xFFFFull);             // stride1 = 0
  g1[7] = 0;
  i32x4 z4 = {0, 0, 0, 0};
  i32x8 z8 = {0, 0, 0, 0, 0, 0, 0, 0};
  __builtin_amdgcn_tensor_load_to_lds(g0, g1, z4, z4, z8, 0);
}

static __device__ __forceinline__ unsigned lds_off(const void* p) {
  // LDS aperture: low 32 bits of a generic shared pointer are the LDS offset
  return (unsigned)(size_t)p;
}

// ---------------------------------------------------------------------------
// Fused flash attention, producer/consumer wave split.
//   grid = (NQ/64, B), block = 256 (8 waves).
//   Waves 0..3 (S-waves): 16 q-rows each; QK^T (full 256 contraction, Q held
//     in regs), online softmax, publish P(f16) + per-row corr to LDS, then
//     PV for d 0..127.
//   Waves 4..7 (helpers): rescale their accumulators with the published corr
//     and do PV for d 128..255 from the published P.
//   K tiles [32x256] f16 and V^T tiles [256x32] f16 double-buffered via TDM.
// ---------------------------------------------------------------------------
__global__ void attn_kernel(const _Float16* __restrict__ qh,
                            const _Float16* __restrict__ kh,
                            const _Float16* __restrict__ vth,
                            float* __restrict__ out) {
  __shared__ _Float16 Kt[2][32][DDIM];    // 2 x 16 KB
  __shared__ _Float16 Vt[2][DDIM][32];    // 2 x 16 KB
  __shared__ _Float16 Pst[4][16][32];     // 4 KB  (per row-group P tile)
  __shared__ float    Cst[4][16];         // per-row correction, this tile
  __shared__ float    Lst[4][16];         // per-row final sum (epilogue)

  const int b    = blockIdx.y;
  const int qt   = blockIdx.x;
  const int wave = threadIdx.x >> 5;
  const int lane = threadIdx.x & 31;
  const int lo   = lane & 15;
  const int hiW  = (lane >= 16) ? 1 : 0;
  const int h8   = hiW ? 8 : 0;
  const int rg   = wave & 3;              // row group
  const bool swv = wave < 4;              // S-wave?
  const int jb   = swv ? 0 : 8;           // this wave's d-tile base
  const float scale = 0.0625f;            // 1/sqrt(256)

  // --- S-waves hold their Q rows in A-operand registers for the whole loop
  HU16 qa[8];
  if (swv) {
    const _Float16* qp =
        qh + ((size_t)b * NQS + qt * 64 + rg * 16 + lo) * DDIM;
    if (lane == 0) __builtin_prefetch(qp, 0, 1);
#pragma unroll
    for (int kk = 0; kk < 8; ++kk) {
      qa[kk].h[0] = *(const v8h*)(qp + kk * 32 + (hiW ? 16 : 0));
      qa[kk].h[1] = *(const v8h*)(qp + kk * 32 + (hiW ? 24 : 8));
    }
  }

  v8f o[8];
#pragma unroll
  for (int j = 0; j < 8; ++j) o[j] = (v8f){};
  float mrow[8], lrow[8];
#pragma unroll
  for (int i = 0; i < 8; ++i) { mrow[i] = -3.0e38f; lrow[i] = 0.0f; }

  const unsigned long long kbase =
      (unsigned long long)(size_t)(kh + (size_t)b * NKS * DDIM);
  const unsigned long long vbase =
      (unsigned long long)(size_t)(vth + (size_t)b * DDIM * NKS);

  // preload tile 0 (wave 0 drives the TDM; EXEC is ignored by tensor ops)
  if (wave == 0) {
    tdm_load_2d(lds_off(&Kt[0][0][0]), kbase, DDIM, 32, DDIM, NKS, DDIM);
    tdm_load_2d(lds_off(&Vt[0][0][0]), vbase, 32, DDIM, NKS, DDIM, NKS);
  }

  for (int t = 0; t < NKS / 32; ++t) {
    if (wave == 0) {
      if (t + 1 < NKS / 32) {
        const int nb = (t + 1) & 1;
        tdm_load_2d(lds_off(&Kt[nb][0][0]),
                    kbase + (unsigned long long)(t + 1) * 32 * DDIM * 2,
                    DDIM, 32, DDIM, NKS, DDIM);
        tdm_load_2d(lds_off(&Vt[nb][0][0]),
                    vbase + (unsigned long long)(t + 1) * 32 * 2,
                    32, DDIM, NKS, DDIM, NKS);
        __builtin_amdgcn_s_wait_tensorcnt(2);   // tile t fully landed in LDS
      } else {
        __builtin_amdgcn_s_wait_tensorcnt(0);
      }
    }
    __syncthreads();                      // (1) tile t visible to all waves
    const int buf = t & 1;

    if (swv) {
      // ---- S = Q * K^T (two 16-col n-tiles), K operands double-buffered
      v8f s0 = (v8f){}, s1 = (v8f){};
      const int co = hiW ? 16 : 0;
      v16h b0c, b1c, b0n, b1n;
      b0c = *(const v16h*)&Kt[buf][lo][co];
      b1c = *(const v16h*)&Kt[buf][lo + 16][co];
#pragma unroll
      for (int kk = 0; kk < 8; ++kk) {
        if (kk < 7) {
          b0n = *(const v16h*)&Kt[buf][lo][(kk + 1) * 32 + co];
          b1n = *(const v16h*)&Kt[buf][lo + 16][(kk + 1) * 32 + co];
        }
        s0 = wmma16(qa[kk].v, b0c, s0);
        s1 = wmma16(qa[kk].v, b1c, s1);
        if (kk < 7) { b0c = b0n; b1c = b1n; }
      }
      // ---- online softmax (row stats reduced across each 16-lane half)
#pragma unroll
      for (int i = 0; i < 8; ++i) {
        float v0 = s0[i] * scale, v1 = s1[i] * scale;
        float mx = fmaxf(v0, v1);
#pragma unroll
        for (int off = 8; off >= 1; off >>= 1)
          mx = fmaxf(mx, __shfl_xor(mx, off, 32));
        const float mnew = fmaxf(mrow[i], mx);
        const float corr = __expf(mrow[i] - mnew);
        const float p0 = __expf(v0 - mnew);
        const float p1 = __expf(v1 - mnew);
        float rs = p0 + p1;
#pragma unroll
        for (int off = 8; off >= 1; off >>= 1) rs += __shfl_xor(rs, off, 32);
        lrow[i] = lrow[i] * corr + rs;
        mrow[i] = mnew;
#pragma unroll
        for (int j = 0; j < 8; ++j) o[j][i] *= corr;
        Pst[rg][i + h8][lo]      = (_Float16)p0;  // C-layout -> row-major P
        Pst[rg][i + h8][lo + 16] = (_Float16)p1;
        if (lo == 0) Cst[rg][i + h8] = corr;      // lanes 0 and 16 write
      }
    }
    __syncthreads();                      // (2) P and corr published

    if (!swv) {
      // helpers rescale their accumulators with this tile's correction
#pragma unroll
      for (int i = 0; i < 8; ++i) {
        const float corr = Cst[rg][i + h8];
#pragma unroll
        for (int j = 0; j < 8; ++j) o[j][i] *= corr;
      }
    }

    // ---- O += P * V over this wave's 8 d-tiles (V double-buffered)
    HU16 pu;
    pu.h[0] = *(const v8h*)&Pst[rg][lo][hiW ? 16 : 0];
    pu.h[1] = *(const v8h*)&Pst[rg][lo][hiW ? 24 : 8];
    {
      const int co = hiW ? 16 : 0;
      v16h bvc, bvn;
      bvc = *(const v16h*)&Vt[buf][jb * 16 + lo][co];
#pragma unroll
      for (int j = 0; j < 8; ++j) {
        if (j < 7)
          bvn = *(const v16h*)&Vt[buf][(jb + j + 1) * 16 + lo][co];
        o[j] = wmma16(pu.v, bvc, o[j]);
        if (j < 7) bvc = bvn;
      }
    }
    __syncthreads();                      // (3) buffers free for next tile
  }

  // ---- epilogue: publish row sums, then out = O / l  (f32 output)
  if (swv && lo == 0) {
#pragma unroll
    for (int i = 0; i < 8; ++i) Lst[rg][i + h8] = lrow[i];
  }
  __syncthreads();

  float* op = out + ((size_t)b * NQS + qt * 64 + rg * 16) * DDIM;
#pragma unroll
  for (int i = 0; i < 8; ++i) {
    const float l   = swv ? lrow[i] : Lst[rg][i + h8];
    const float inv = 1.0f / l;
#pragma unroll
    for (int j = 0; j < 8; ++j)
      op[(size_t)(i + h8) * DDIM + (jb + j) * 16 + lo] = o[j][i] * inv;
  }
}

// ---------------------------------------------------------------------------
// Host-side orchestration
// ---------------------------------------------------------------------------
extern "C" void kernel_launch(void* const* d_in, const int* in_sizes, int n_in,
                              void* d_out, int out_size, void* d_ws,
                              size_t ws_size, hipStream_t stream) {
  const float* query = (const float*)d_in[0];   // [8,2048,256]
  const float* key   = (const float*)d_in[1];   // [8,2048,256]
  const float* Wq    = (const float*)d_in[2];   // [256,256]
  const float* bq    = (const float*)d_in[3];   // [256]
  const float* Wk    = (const float*)d_in[4];   // [256,256]
  const float* bk    = (const float*)d_in[5];   // [256]
  float* out = (float*)d_out;

  char* ws = (char*)d_ws;
  size_t off = 0;
  _Float16* wq_h  = (_Float16*)(ws + off); off += (size_t)DDIM * DDIM * 2;
  _Float16* wk_h  = (_Float16*)(ws + off); off += (size_t)DDIM * DDIM * 2;
  _Float16* key_h = (_Float16*)(ws + off); off += (size_t)BB * NKS * DDIM * 2;
  _Float16* qf_h  = (_Float16*)(ws + off); off += (size_t)BB * NQS * DDIM * 2;
  _Float16* q_h   = (_Float16*)(ws + off); off += (size_t)BB * NQS * DDIM * 2;
  _Float16* k_h   = (_Float16*)(ws + off); off += (size_t)BB * NKS * DDIM * 2;
  _Float16* vt_h  = (_Float16*)(ws + off); off += (size_t)BB * DDIM * NKS * 2;

  // 1) precision converts
  cvt_f32_f16_kernel<<<(DDIM * DDIM + 255) / 256, 256, 0, stream>>>(
      Wq, wq_h, DDIM * DDIM);
  cvt_f32_f16_kernel<<<(DDIM * DDIM + 255) / 256, 256, 0, stream>>>(
      Wk, wk_h, DDIM * DDIM);
  cvt_f32_f16_kernel<<<(BB * NKS * DDIM + 255) / 256, 256, 0, stream>>>(
      key, key_h, BB * NKS * DDIM);

  // 2) real(FFT) over seq axis as cosine-DFT WMMA GEMM
  dft_kernel<<<dim3(NQS / 64, BB), 256, 0, stream>>>(query, qf_h);

  // 3) projections (WMMA GEMMs)
  proj_kernel<<<dim3(BB * NQS / 64), 256, 0, stream>>>(qf_h, wq_h, bq, q_h);
  proj_kernel<<<dim3(BB * NKS / 64), 256, 0, stream>>>(key_h, wk_h, bk, k_h);

  // 4) V^T for contiguous PV B-operands
  transpose_v_kernel<<<dim3(NKS / 32, DDIM / 32, BB), dim3(32, 32), 0,
                       stream>>>(key_h, vt_h);

  // 5) fused flash attention with TDM double buffering
  attn_kernel<<<dim3(NQS / 64, BB), 256, 0, stream>>>(q_h, k_h, vt_h, out);
}
